// QuadraticHeterogenousCrossAttention_18373870092459
// MI455X (gfx1250) — matrix operand order repaired and checked by reading, not compiled, not run on hardware
//
#include <hip/hip_runtime.h>
#include <hip/hip_bf16.h>

typedef __attribute__((ext_vector_type(16))) __bf16 v16bf;
typedef __attribute__((ext_vector_type(8)))  __bf16 v8bf;
typedef __attribute__((ext_vector_type(8)))  float  v8f;

// ---------- helpers ----------

__device__ __forceinline__ __bf16 f2bf(float f) {
    return (__bf16)f;   // native fptrunc -> hardware bf16 cvt on gfx1250
}

__device__ __forceinline__ v8f zero8() {
    v8f z;
#pragma unroll
    for (int i = 0; i < 8; ++i) z[i] = 0.0f;
    return z;
}

__device__ __forceinline__ v16bf cat16(v8bf lo, v8bf hi) {
    v16bf r;
#pragma unroll
    for (int i = 0; i < 8; ++i) { r[i] = lo[i]; r[i + 8] = hi[i]; }
    return r;
}

// A-operand (16x32, per-lane K = e + 2*off8) from a bf16 row pointer
__device__ __forceinline__ v16bf load_a_bf16(const __bf16* rowbase, int off8) {
    v8bf lo = *(const v8bf*)(rowbase + 2 * off8);
    v8bf hi = *(const v8bf*)(rowbase + 2 * off8 + 8);
    return cat16(lo, hi);
}

// B-operand (32x16, per-lane K = e + off16) from a bf16 column pointer
__device__ __forceinline__ v16bf load_b_bf16(const __bf16* colk, int off16) {
    v8bf lo = *(const v8bf*)(colk + off16);
    v8bf hi = *(const v8bf*)(colk + off16 + 8);
    return cat16(lo, hi);
}

__device__ __forceinline__ v8f wmma_bf16(v16bf a, v16bf b, v8f c) {
    // v_wmma_f32_16x16x32_bf16
    return __builtin_amdgcn_wmma_f32_16x16x32_bf16(false, a, false, b, (short)0, c,
                                                   false, false);
}

// ---------- constants ----------
// B=2, S=2048 per batch, D=256, H=8, dh=32, F=4
#define S_LEN   2048
#define D_DIM   256
#define BH_CNT  16
#define FSTRIDE ((size_t)BH_CNT * S_LEN * 32)   // elems per feature buffer
#define WELEMS  ((size_t)D_DIM * D_DIM)         // 65536 elems per weight matrix

// ---------------------------------------------------------------------------
// Kernel 0: fp32 -> bf16 bulk convert (one-time; makes all GEMMs pure bf16)
// ---------------------------------------------------------------------------
__global__ __launch_bounds__(256) void cvt_kernel(
    const float* __restrict__ src, __bf16* __restrict__ dst, int n)
{
    const int i = (blockIdx.x * 256 + threadIdx.x) * 8;
    if (i + 8 <= n) {
        const float4 f0 = *(const float4*)(src + i);
        const float4 f1 = *(const float4*)(src + i + 4);
        v8bf o;
        o[0] = f2bf(f0.x); o[1] = f2bf(f0.y); o[2] = f2bf(f0.z); o[3] = f2bf(f0.w);
        o[4] = f2bf(f1.x); o[5] = f2bf(f1.y); o[6] = f2bf(f1.z); o[7] = f2bf(f1.w);
        *(v8bf*)(dst + i) = o;
    }
}

// ---------------------------------------------------------------------------
// Kernel 1: projection  Y = Xbf @ Wbf^T + bias  (+ optional RoPE, + scale)
// grid (256 mtiles, 16 ntiles), block 32 (one wave per 16x16 tile)
// transposed==0: store bf16 [bh][s][dh]   (Q / K layout)
// transposed==1: store bf16 [bh][dh][s]   (V^T layout)
// ---------------------------------------------------------------------------
__global__ __launch_bounds__(32) void proj_kernel(
    const __bf16* __restrict__ X, const __bf16* __restrict__ W,
    const float* __restrict__ bias, __bf16* __restrict__ dst,
    int rope, float outScale, int transposed)
{
    const int lane  = threadIdx.x & 31;
    const int mt    = blockIdx.x;
    const int nt    = blockIdx.y;
    const int mrow  = lane & 15;
    const int off8  = (lane < 16) ? 0 : 8;
    const int off16 = (lane < 16) ? 0 : 16;

    const __bf16* xrow = X + (size_t)(mt * 16 + mrow) * D_DIM;
    const __bf16* wrow = W + (size_t)(nt * 16 + mrow) * D_DIM;

    v8f acc = zero8();
#pragma unroll
    for (int kc = 0; kc < D_DIM; kc += 32) {
        v16bf a = load_a_bf16(xrow + kc, off8);
        v16bf b = load_b_bf16(wrow + kc, off16);
        acc = wmma_bf16(a, b, acc);
    }

    const int n    = lane & 15;
    const int dout = nt * 16 + n;
    const float bval = bias[dout];
    const int mAdd = (lane < 16) ? 0 : 8;

    // RoPE angle pieces that depend only on the lane (dout)
    const int   dh   = dout & 31;
    const int   j    = dh >> 1;
    const float freq = __powf(1000.0f, -(float)j * (1.0f / 16.0f));

#pragma unroll
    for (int r = 0; r < 8; ++r) {
        const int m = r + mAdd;
        const int R = mt * 16 + m;          // global row 0..4095
        float v = acc[r] + bval;
        if (rope) {
            const int s   = R & (S_LEN - 1);
            const float pos = (float)(s >> 8);     // repeat(arange(T=8), 256)
            float sn, cs;
            __sincosf(pos * freq, &sn, &cs);
            const float pv = __shfl_xor(v, 1, 32); // partner dout^1, same row
            v = (dout & 1) ? (pv * sn + v * cs) : (v * cs - pv * sn);
        }
        v *= outScale;
        const int b  = R >> 11;
        const int s  = R & (S_LEN - 1);
        const int bh = b * 8 + (dout >> 5);
        if (!transposed) {
            dst[((size_t)bh * S_LEN + s) * 32 + (dout & 31)] = f2bf(v);
        } else {
            dst[((size_t)bh * 32 + (dout & 31)) * S_LEN + s] = f2bf(v);
        }
    }
}

// ---------------------------------------------------------------------------
// Kernel 2: gated 4-feature flash attention (transposed-score formulation)
// grid (16, 1, 16 bh), block 256 (8 waves; each wave = one 16-query tile)
// oa: bf16 [bh][s(2048)][dh(32)]  (head-concat layout for the output GEMM)
// ---------------------------------------------------------------------------
__global__ __launch_bounds__(256) void attn_kernel(
    const __bf16* __restrict__ qp, const __bf16* __restrict__ kp,
    const __bf16* __restrict__ vt, const float* __restrict__ fw,
    __bf16* __restrict__ oa)
{
    __shared__ alignas(16) __bf16 sK[32 * 32];   // [key_local][dh]
    __shared__ alignas(16) __bf16 sV[32 * 32];   // [dh][key_local]

    const int tid   = threadIdx.x;
    const int lane  = tid & 31;
    const int wave  = tid >> 5;
    const int bh    = blockIdx.z;
    const int q0    = (blockIdx.x * 8 + wave) * 16;
    const int n     = lane & 15;
    const int off8  = (lane < 16) ? 0 : 8;
    const int off16 = (lane < 16) ? 0 : 16;

    // gates = softmax(feature_weights)
    float g[4];
    {
        const float w0 = fw[0], w1 = fw[1], w2 = fw[2], w3 = fw[3];
        const float mm = fmaxf(fmaxf(w0, w1), fmaxf(w2, w3));
        const float e0 = __expf(w0 - mm), e1 = __expf(w1 - mm);
        const float e2 = __expf(w2 - mm), e3 = __expf(w3 - mm);
        const float si = 1.0f / (e0 + e1 + e2 + e3);
        g[0] = e0 * si; g[1] = e1 * si; g[2] = e2 * si; g[3] = e3 * si;
    }

    // Q^T B-operand: col n = query q0+n, K-dim = dh (1/sqrt(dh) already folded)
    const __bf16* qptr = qp + ((size_t)bh * S_LEN + q0 + n) * 32 + off16;
    const v16bf Bq = cat16(*(const v8bf*)qptr, *(const v8bf*)(qptr + 8));

    v8f total0 = zero8();   // out^T tile, dh rows 0..15
    v8f total1 = zero8();   // out^T tile, dh rows 16..31

    for (int f = 0; f < 4; ++f) {
        const __bf16* kbase = kp + (size_t)f * FSTRIDE + (size_t)bh * S_LEN * 32;
        const __bf16* vbase = vt + (size_t)f * FSTRIDE + (size_t)bh * 32 * S_LEN;

        v8f acc0 = zero8(), acc1 = zero8();
        float mrun = -3.0e38f, lrun = 0.0f;

        for (int c = 0; c < S_LEN; c += 32) {
            __syncthreads();
            // cooperative 4KB stage of K-chunk and V^T-chunk into LDS
            if (tid < 128) {
                const int row = tid >> 2, qq = tid & 3;
                *(v8bf*)(sK + row * 32 + qq * 8) =
                    *(const v8bf*)(kbase + (size_t)(c + row) * 32 + qq * 8);
            } else {
                const int tt = tid - 128;
                const int d = tt >> 2, qq = tt & 3;
                *(v8bf*)(sV + d * 32 + qq * 8) =
                    *(const v8bf*)(vbase + (size_t)d * S_LEN + c + qq * 8);
            }
            __syncthreads();
            if (tid == 0 && c + 32 < S_LEN)       // hint next chunk toward L2/L0
                __builtin_prefetch(kbase + (size_t)(c + 32) * 32, 0, 1);

            // scores^T tiles: (key x query) = K_tile(16x32dh) @ Q^T(32dh x 16)
            const v16bf a0 = load_a_bf16(sK + (lane & 15) * 32, off8);
            const v16bf a1 = load_a_bf16(sK + (16 + (lane & 15)) * 32, off8);
            v8f S0 = wmma_bf16(a0, Bq, zero8());
            v8f S1 = wmma_bf16(a1, Bq, zero8());

            // online softmax over the 32-key chunk (per query column)
            float mc = -3.0e38f;
#pragma unroll
            for (int r = 0; r < 8; ++r) mc = fmaxf(mc, fmaxf(S0[r], S1[r]));
            mc = fmaxf(mc, __shfl_xor(mc, 16, 32));
            const float mnew  = fmaxf(mrun, mc);
            const float alpha = __expf(mrun - mnew);

            float p0[8], p1[8];
            float ssum = 0.0f;
#pragma unroll
            for (int r = 0; r < 8; ++r) {
                p0[r] = __expf(S0[r] - mnew);
                p1[r] = __expf(S1[r] - mnew);
                ssum += p0[r] + p1[r];
            }
            ssum += __shfl_xor(ssum, 16, 32);
            lrun = lrun * alpha + ssum;
            mrun = mnew;
#pragma unroll
            for (int r = 0; r < 8; ++r) { acc0[r] *= alpha; acc1[r] *= alpha; }

            // rebuild P (32key x 16query) in the WMMA B layout via lane^16 swaps
            v16bf bp;
#pragma unroll
            for (int r = 0; r < 8; ++r) {
                const float sh0 = __shfl_xor(p0[r], 16, 32);
                const float sh1 = __shfl_xor(p1[r], 16, 32);
                bp[r]     = f2bf((lane < 16) ? p0[r] : sh1);
                bp[r + 8] = f2bf((lane < 16) ? sh0 : p1[r]);
            }

            // out^T += V^T(16dh x 32key) @ P(32key x 16query)
            const v16bf av0 = load_a_bf16(sV + (lane & 15) * 32, off8);
            const v16bf av1 = load_a_bf16(sV + (16 + (lane & 15)) * 32, off8);
            acc0 = wmma_bf16(av0, bp, acc0);
            acc1 = wmma_bf16(av1, bp, acc1);
        }

        const float inv = g[f] / lrun;
#pragma unroll
        for (int r = 0; r < 8; ++r) {
            total0[r] += acc0[r] * inv;
            total1[r] += acc1[r] * inv;
        }
    }

    // scatter out^T tile into head-concat [bh][s][dh] bf16 layout (once)
    __bf16* obase = oa + (size_t)bh * S_LEN * 32;
    const int mAdd = (lane < 16) ? 0 : 8;
#pragma unroll
    for (int r = 0; r < 8; ++r) {
        obase[(size_t)(q0 + n) * 32 + (r + mAdd)]      = f2bf(total0[r]);
        obase[(size_t)(q0 + n) * 32 + 16 + (r + mAdd)] = f2bf(total1[r]);
    }
}

// ---------------------------------------------------------------------------
// Kernel 3: output projection  o = concat_heads(out) @ Wo^T + bo   (fp32 out)
// grid (256 mtiles, 16 ntiles), block 32; pure bf16 operands
// ---------------------------------------------------------------------------
__global__ __launch_bounds__(32) void outproj_kernel(
    const __bf16* __restrict__ oa, const __bf16* __restrict__ WoBf,
    const float* __restrict__ bo, float* __restrict__ out)
{
    const int lane  = threadIdx.x & 31;
    const int mt    = blockIdx.x;
    const int nt    = blockIdx.y;
    const int mrow  = lane & 15;
    const int off8  = (lane < 16) ? 0 : 8;
    const int off16 = (lane < 16) ? 0 : 16;

    const int b     = (mt * 16) >> 11;
    const int sbase = (mt * 16) & (S_LEN - 1);

    v8f acc = zero8();
#pragma unroll
    for (int h = 0; h < 8; ++h) {
        // A rows: out-concat row s, K-dim = dh of head h  (contiguous bf16)
        const __bf16* arow =
            oa + ((size_t)(b * 8 + h) * S_LEN + sbase + mrow) * 32;
        const v16bf a = load_a_bf16(arow, off8);
        const v16bf bm =
            load_b_bf16(WoBf + (size_t)(nt * 16 + mrow) * D_DIM + h * 32, off16);
        acc = wmma_bf16(a, bm, acc);
    }

    const int n    = lane & 15;
    const float bv = bo[nt * 16 + n];
    const int mAdd = (lane < 16) ? 0 : 8;
#pragma unroll
    for (int r = 0; r < 8; ++r) {
        const int R = mt * 16 + r + mAdd;
        out[(size_t)R * D_DIM + nt * 16 + n] = acc[r] + bv;
    }
}

// ---------------------------------------------------------------------------
extern "C" void kernel_launch(void* const* d_in, const int* in_sizes, int n_in,
                              void* d_out, int out_size, void* d_ws, size_t ws_size,
                              hipStream_t stream)
{
    const float* q_data = (const float*)d_in[0];
    const float* feats[4] = {(const float*)d_in[1], (const float*)d_in[2],
                             (const float*)d_in[3], (const float*)d_in[4]};
    const float* Wq = (const float*)d_in[5];
    const float* bq = (const float*)d_in[6];
    const float* Wk = (const float*)d_in[7];
    const float* bk = (const float*)d_in[8];
    const float* Wv = (const float*)d_in[9];
    const float* bv = (const float*)d_in[10];
    const float* Wo = (const float*)d_in[11];
    const float* bo = (const float*)d_in[12];
    const float* fw = (const float*)d_in[13];

    // workspace (bytes): qp 2MB | kp 8MB | vt 8MB | oa 2MB | Xbf 10MB | Wbf 1.25MB
    char* ws = (char*)d_ws;
    __bf16* qp  = (__bf16*)(ws);
    __bf16* kp  = (__bf16*)(ws + ((size_t)2 << 20));
    __bf16* vt  = (__bf16*)(ws + ((size_t)10 << 20));
    __bf16* oa  = (__bf16*)(ws + ((size_t)18 << 20));
    __bf16* Xbf = (__bf16*)(ws + ((size_t)20 << 20));
    __bf16* Wbf = (__bf16*)(ws + ((size_t)30 << 20));

    const int NX = 4096 * 256;                 // elems per activation tensor
    __bf16* Xq   = Xbf;                        // q_data
    __bf16* Xf[4] = {Xbf + (size_t)1 * NX, Xbf + (size_t)2 * NX,
                     Xbf + (size_t)3 * NX, Xbf + (size_t)4 * NX};
    __bf16* WqB  = Wbf + 0 * WELEMS;
    __bf16* WkB  = Wbf + 1 * WELEMS;           // 4 matrices
    __bf16* WvB  = Wbf + 5 * WELEMS;           // 4 matrices
    __bf16* WoB  = Wbf + 9 * WELEMS;

    // one-time fp32 -> bf16 conversion (memory-bound, ~25MB total)
    cvt_kernel<<<NX / 2048, 256, 0, stream>>>(q_data, Xq, NX);
    for (int i = 0; i < 4; ++i)
        cvt_kernel<<<NX / 2048, 256, 0, stream>>>(feats[i], Xf[i], NX);
    cvt_kernel<<<32, 256, 0, stream>>>(Wq, WqB, (int)WELEMS);
    for (int i = 0; i < 4; ++i) {
        cvt_kernel<<<32, 256, 0, stream>>>(Wk + i * WELEMS, WkB + i * WELEMS,
                                           (int)WELEMS);
        cvt_kernel<<<32, 256, 0, stream>>>(Wv + i * WELEMS, WvB + i * WELEMS,
                                           (int)WELEMS);
    }
    cvt_kernel<<<32, 256, 0, stream>>>(Wo, WoB, (int)WELEMS);

    const float qscale = 0.17677669529663687f;  // 1/sqrt(dh=32), folded into Q

    dim3 pg(256, 16), pb(32);
    proj_kernel<<<pg, pb, 0, stream>>>(Xq, WqB, bq, qp, /*rope=*/1, qscale, 0);
    for (int i = 0; i < 4; ++i) {
        proj_kernel<<<pg, pb, 0, stream>>>(Xf[i], WkB + i * WELEMS, bk + i * 256,
                                           kp + (size_t)i * FSTRIDE,
                                           /*rope=*/1, 1.0f, 0);
        proj_kernel<<<pg, pb, 0, stream>>>(Xf[i], WvB + i * WELEMS, bv + i * 256,
                                           vt + (size_t)i * FSTRIDE,
                                           /*rope=*/0, 1.0f, /*transposed=*/1);
    }
    attn_kernel<<<dim3(16, 1, 16), 256, 0, stream>>>(qp, kp, vt, fw, oa);
    outproj_kernel<<<dim3(256, 16), 32, 0, stream>>>(oa, WoB, bo, (float*)d_out);
}
